// distGCN_74929999446102
// MI455X (gfx1250) — compile-verified, matrix operand order checked by reading, not run on hardware
//
#include <hip/hip_runtime.h>

typedef __attribute__((ext_vector_type(16))) __bf16 v16bf;
typedef __attribute__((ext_vector_type(8)))  float  v8f;

#define LN_EPS 1e-5f

__device__ __forceinline__ unsigned short f2bf_bits(float f) {
    union { float f; unsigned u; } x; x.f = f;
    unsigned r = x.u + 0x7FFFu + ((x.u >> 16) & 1u);   // round-to-nearest-even
    return (unsigned short)(r >> 16);
}

// ---------------------------------------------------------------- degrees ---
__global__ void k_degrees(const int* __restrict__ src, const int* __restrict__ dst,
                          float* __restrict__ deg_out, float* __restrict__ deg_in, int E) {
    int e = blockIdx.x * blockDim.x + threadIdx.x;
    if (e < E) {
        atomicAdd(&deg_out[src[e]], 1.0f);
        atomicAdd(&deg_in [dst[e]], 1.0f);
    }
}

__global__ void k_norms(float* __restrict__ a, float* __restrict__ b, int n) {
    int i = blockIdx.x * blockDim.x + threadIdx.x;
    if (i < n) {
        a[i] = rsqrtf(fmaxf(a[i], 1.0f));
        b[i] = rsqrtf(fmaxf(b[i], 1.0f));
    }
}

// ------------------------------------- W -> bf16, WMMA-fragment swizzled ---
// Wsw[((tile*4 + kb)*32 + lane)*16 + e] = bf16( W[K][col] ) with
//   koff = (lane<16)?0:8 ; K = kb*32 + koff + e + (e>=8 ? 8 : 0)
//   col  = tile*16 + (lane&15)
__global__ void k_pack_w(const float* __restrict__ W, unsigned short* __restrict__ Wsw, int nc) {
    int i = blockIdx.x * blockDim.x + threadIdx.x;
    if (i >= 128 * nc) return;
    int e    = i & 15;
    int lane = (i >> 4) & 31;
    int kb   = (i >> 9) & 3;
    int tile = i >> 11;
    int koff = (lane < 16) ? 0 : 8;
    int K    = kb * 32 + koff + e + ((e >= 8) ? 8 : 0);
    int col  = tile * 16 + (lane & 15);
    Wsw[i] = f2bf_bits(W[K * nc + col]);
}

// ------------------------------------------------------- edge aggregation ---
// One wave32 per edge, 4 floats per lane (d = 128).
__global__ __launch_bounds__(256)
void k_aggregate(const float* __restrict__ h, const int* __restrict__ src,
                 const int* __restrict__ dst, const float* __restrict__ norm_o,
                 float* __restrict__ agg, int E) {
    long long t = (long long)blockIdx.x * blockDim.x + threadIdx.x;
    int e    = (int)(t >> 5);
    int lane = (int)(t & 31);
    if (e >= E) return;
    int s = src[e], d = dst[e];
    float w = norm_o[s];
    const float4 v = *reinterpret_cast<const float4*>(h + (size_t)s * 128 + lane * 4);
    float* dp = agg + (size_t)d * 128 + lane * 4;
    atomicAdd(dp + 0, v.x * w);
    atomicAdd(dp + 1, v.y * w);
    atomicAdd(dp + 2, v.z * w);
    atomicAdd(dp + 3, v.w * w);
}

// ----------------- A = agg * norm_i -> bf16, WMMA-fragment swizzled --------
// One thread per 8-element fragment chunk (16 bytes out).
// Element order matches W swizzle: per 16-row tile t,
//   Abf[t*2048 + (kb*32 + lane)*16 + e]  (e handled 8 at a time via `half`)
__global__ void k_scale_pack_a(const float* __restrict__ agg, const float* __restrict__ norm_i,
                               unsigned short* __restrict__ Abf, int nrows, int ngroups) {
    int i = blockIdx.x * blockDim.x + threadIdx.x;
    if (i >= ngroups) return;
    int local = i & 255;           // group within 16-row tile (256 groups = 2048 elems)
    int t     = i >> 8;            // 16-row tile
    int half  = local & 1;         // e in [0,8) or [8,16)
    int lane  = (local >> 1) & 31;
    int kb    = local >> 6;
    int row   = t * 16 + (lane & 15);
    int koff  = (lane < 16) ? 0 : 8;
    int Kb    = kb * 32 + koff + (half ? 16 : 0);
    uint4 o = {0u, 0u, 0u, 0u};
    if (row < nrows) {
        float w = norm_i[row];
        const float* s = agg + (size_t)row * 128 + Kb;
        float4 a = *reinterpret_cast<const float4*>(s);
        float4 b = *reinterpret_cast<const float4*>(s + 4);
        o.x = (unsigned)f2bf_bits(a.x * w) | ((unsigned)f2bf_bits(a.y * w) << 16);
        o.y = (unsigned)f2bf_bits(a.z * w) | ((unsigned)f2bf_bits(a.w * w) << 16);
        o.z = (unsigned)f2bf_bits(b.x * w) | ((unsigned)f2bf_bits(b.y * w) << 16);
        o.w = (unsigned)f2bf_bits(b.z * w) | ((unsigned)f2bf_bits(b.w * w) << 16);
    }
    *reinterpret_cast<uint4*>(Abf + (size_t)i * 8) = o;
}

// ------------------------------------------- fused GEMM(+bias,LN,relu) ------
// A tile (8 KB, fragment-swizzled bf16) arrives via global_load_async_to_lds;
// B fragments loaded pre-swizzled from global; v_wmma_f32_16x16x32_bf16.
template <int NC, bool LN>
__global__ __launch_bounds__(256)
void k_gcn_gemm(const unsigned short* __restrict__ Abf,
                const unsigned short* __restrict__ Wsw,
                const float* __restrict__ bias,
                const float* __restrict__ gamma, const float* __restrict__ beta,
                float* __restrict__ out, int nrows) {
    __shared__ v16bf Asw[2][4][32];          // 8 KB: [m-subtile][kb][lane] fragments
    __shared__ float Cs[32][128];            // 16 KB f32 staging for LN

    const int tid  = threadIdx.x;
    const int wave = tid >> 5;
    const int lane = tid & 31;
    const int row0 = blockIdx.x * 32;

    __builtin_prefetch(Wsw, 0, 3);           // global_prefetch_b8

    // Async global->LDS copy of this block's contiguous 8 KB A tile.
    {
        unsigned ldsbase = (unsigned)(uintptr_t)(&Asw[0][0][0]);
        const char* gsrc = (const char*)Abf + (size_t)blockIdx.x * 8192;
        unsigned off0 = (unsigned)tid * 16u;
        asm volatile("global_load_async_to_lds_b128 %0, %1, %2"
                     :: "v"(ldsbase + off0), "v"(off0), "s"(gsrc) : "memory");
        unsigned off1 = off0 + 4096u;
        asm volatile("global_load_async_to_lds_b128 %0, %1, %2"
                     :: "v"(ldsbase + off1), "v"(off1), "s"(gsrc) : "memory");
    }

    // B fragments: aligned 32B vector loads (overlap with the async copy)
    v16bf bfrag[4];
    if (wave < NC / 16) {
        const v16bf* Wv = reinterpret_cast<const v16bf*>(Wsw);
#pragma unroll
        for (int kb = 0; kb < 4; ++kb)
            bfrag[kb] = Wv[(wave * 4 + kb) * 32 + lane];
    }

    asm volatile("s_wait_asynccnt 0x0" ::: "memory");
    __syncthreads();

    v8f acc[2] = {};
    const int nsub = lane & 15;
    const int roff = (lane < 16) ? 0 : 8;

    if (wave < NC / 16) {
        const int col = wave * 16 + nsub;

        // Preload all 8 A fragments (bank-conflict-free 32B/lane reads),
        // then run the 8 WMMAs back-to-back.
        v16bf af[2][4];
#pragma unroll
        for (int mt = 0; mt < 2; ++mt)
#pragma unroll
            for (int kb = 0; kb < 4; ++kb)
                af[mt][kb] = Asw[mt][kb][lane];

#pragma unroll
        for (int mt = 0; mt < 2; ++mt)
#pragma unroll
            for (int kb = 0; kb < 4; ++kb)
                acc[mt] = __builtin_amdgcn_wmma_f32_16x16x32_bf16(
                    false, af[mt][kb], false, bfrag[kb], (short)0, acc[mt], false, false);

        // Pipeline shape for the scheduler: all 16 DS reads first, then the
        // 8 WMMAs, so each WMMA's wait only covers already-covered latency.
        __builtin_amdgcn_sched_group_barrier(0x100, 16, 0);  // DS reads
        __builtin_amdgcn_sched_group_barrier(0x008,  8, 0);  // WMMA group

        if (LN) {
#pragma unroll
            for (int mt = 0; mt < 2; ++mt)
#pragma unroll
                for (int r = 0; r < 8; ++r)
                    Cs[mt * 16 + roff + r][col] = acc[mt][r] + bias[col];
        } else {
#pragma unroll
            for (int mt = 0; mt < 2; ++mt)
#pragma unroll
                for (int r = 0; r < 8; ++r) {
                    int g = row0 + mt * 16 + roff + r;
                    if (g < nrows) out[(size_t)g * NC + col] = acc[mt][r] + bias[col];
                }
        }
    }

    if (LN) {
        __syncthreads();
        // 8 waves x 4 rows; 128 cols -> 4 per lane; wave32 shuffle reductions
#pragma unroll
        for (int rr = 0; rr < 4; ++rr) {
            int r = wave * 4 + rr;
            int g = row0 + r;
            float v[4];
#pragma unroll
            for (int j = 0; j < 4; ++j) v[j] = Cs[r][lane * 4 + j];
            float s = v[0] + v[1] + v[2] + v[3];
#pragma unroll
            for (int m = 16; m >= 1; m >>= 1) s += __shfl_xor(s, m, 32);
            float mu = s * (1.0f / 128.0f);
            float q = 0.0f;
#pragma unroll
            for (int j = 0; j < 4; ++j) { float d = v[j] - mu; q += d * d; }
#pragma unroll
            for (int m = 16; m >= 1; m >>= 1) q += __shfl_xor(q, m, 32);
            float inv = rsqrtf(q * (1.0f / 128.0f) + LN_EPS);
            if (g < nrows) {
#pragma unroll
                for (int j = 0; j < 4; ++j) {
                    int c = lane * 4 + j;
                    float y = (v[j] - mu) * inv * gamma[c] + beta[c];
                    out[(size_t)g * 128 + c] = fmaxf(y, 0.0f);
                }
            }
        }
    }
}

// ----------------------------------------------------------------- driver ---
static inline char* align_up(char* p, size_t a) {
    return (char*)(((uintptr_t)p + (a - 1)) & ~(uintptr_t)(a - 1));
}

extern "C" void kernel_launch(void* const* d_in, const int* in_sizes, int n_in,
                              void* d_out, int out_size, void* d_ws, size_t ws_size,
                              hipStream_t stream) {
    const float* feats = (const float*)d_in[0];
    const int*   src   = (const int*)d_in[1];
    const int*   dst   = (const int*)d_in[2];
    const float* W0  = (const float*)d_in[3];
    const float* b0  = (const float*)d_in[4];
    const float* g0  = (const float*)d_in[5];
    const float* be0 = (const float*)d_in[6];
    const float* W1  = (const float*)d_in[7];
    const float* b1  = (const float*)d_in[8];
    const float* g1  = (const float*)d_in[9];
    const float* be1 = (const float*)d_in[10];
    const float* W2  = (const float*)d_in[11];
    const float* b2  = (const float*)d_in[12];
    float* out = (float*)d_out;

    const int N = in_sizes[0] / 128;
    const int E = in_sizes[1];
    const int P = ((N + 31) / 32) * 32;      // rows padded to 32

    char* p = (char*)d_ws;
    float* norm_o = (float*)p; p = align_up(p + (size_t)N * 4, 256);
    float* norm_i = (float*)p; p = align_up(p + (size_t)N * 4, 256);
    float* agg    = (float*)p; p = align_up(p + (size_t)N * 128 * 4, 256);
    float* h      = (float*)p; p = align_up(p + (size_t)N * 128 * 4, 256);
    unsigned short* Abf = (unsigned short*)p; p = align_up(p + (size_t)P * 128 * 2, 256);
    unsigned short* W0b = (unsigned short*)p; p = align_up(p + 128 * 128 * 2, 256);
    unsigned short* W1b = (unsigned short*)p; p = align_up(p + 128 * 128 * 2, 256);
    unsigned short* W2b = (unsigned short*)p; p = align_up(p + 128 * 64 * 2, 256);

    // degrees -> norms (recomputed every call; deterministic work)
    hipMemsetAsync(norm_o, 0, (size_t)N * 4, stream);
    hipMemsetAsync(norm_i, 0, (size_t)N * 4, stream);
    k_degrees<<<(E + 255) / 256, 256, 0, stream>>>(src, dst, norm_o, norm_i, E);
    k_norms<<<(N + 255) / 256, 256, 0, stream>>>(norm_o, norm_i, N);

    // pack weights into per-lane WMMA fragment order (bf16)
    k_pack_w<<<(128 * 128 + 255) / 256, 256, 0, stream>>>(W0, W0b, 128);
    k_pack_w<<<(128 * 128 + 255) / 256, 256, 0, stream>>>(W1, W1b, 128);
    k_pack_w<<<(128 * 64 + 255) / 256, 256, 0, stream>>>(W2, W2b, 64);

    const long long aggThreads = (long long)E * 32;
    const int aggBlocks   = (int)((aggThreads + 255) / 256);
    const int gemmBlocks  = P / 32;
    const int packGroups  = P * 16;          // (P/16 tiles) * 256 groups
    const int packBlocks  = (packGroups + 255) / 256;

    // layer 0
    hipMemsetAsync(agg, 0, (size_t)N * 128 * 4, stream);
    k_aggregate<<<aggBlocks, 256, 0, stream>>>(feats, src, dst, norm_o, agg, E);
    k_scale_pack_a<<<packBlocks, 256, 0, stream>>>(agg, norm_i, Abf, N, packGroups);
    k_gcn_gemm<128, true><<<gemmBlocks, 256, 0, stream>>>(Abf, W0b, b0, g0, be0, h, N);

    // layer 1
    hipMemsetAsync(agg, 0, (size_t)N * 128 * 4, stream);
    k_aggregate<<<aggBlocks, 256, 0, stream>>>(h, src, dst, norm_o, agg, E);
    k_scale_pack_a<<<packBlocks, 256, 0, stream>>>(agg, norm_i, Abf, N, packGroups);
    k_gcn_gemm<128, true><<<gemmBlocks, 256, 0, stream>>>(Abf, W1b, b1, g1, be1, h, N);

    // final layer (C = 64, no layernorm)
    hipMemsetAsync(agg, 0, (size_t)N * 128 * 4, stream);
    k_aggregate<<<aggBlocks, 256, 0, stream>>>(h, src, dst, norm_o, agg, E);
    k_scale_pack_a<<<packBlocks, 256, 0, stream>>>(agg, norm_i, Abf, N, packGroups);
    k_gcn_gemm<64, false><<<gemmBlocks, 256, 0, stream>>>(Abf, W2b, b2, nullptr, nullptr, out, N);
}
